// _NonLocalBlockND_42975442764540
// MI455X (gfx1250) — matrix operand order repaired and checked by reading, not compiled, not run on hardware
//
#include <hip/hip_runtime.h>
#include <hip/hip_bf16.h>

#define BN 8
#define CC 256
#define CI 128
#define HH 64
#define WW 64
#define NN 4096   // H*W
#define NP 1024   // pooled pixels
#define EPSV 1e-5f

typedef __bf16 bf16_t;
typedef __attribute__((ext_vector_type(16))) __bf16 v16bf;
typedef __attribute__((ext_vector_type(8)))  __bf16 v8bf;
typedef __attribute__((ext_vector_type(8)))  float  v8f;

__device__ __forceinline__ bf16_t f2bf(float f) {
  unsigned u = __builtin_bit_cast(unsigned, f);
  u += 0x7FFFu + ((u >> 16) & 1u);            // round-to-nearest-even
  unsigned short h = (unsigned short)(u >> 16);
  return __builtin_bit_cast(bf16_t, h);
}
__device__ __forceinline__ float bf2f(bf16_t b) {
  unsigned short h = __builtin_bit_cast(unsigned short, b);
  unsigned u = ((unsigned)h) << 16;
  return __builtin_bit_cast(float, u);
}

// ---------------------------------------------------------------------------
// Wave-level WMMA GEMM core: one wave computes a (MT*16) x (NT*16) f32 tile
// of D = A * B.  A row-major [M,K] bf16, B column-major (Bc[col*K + k]) bf16.
// Fragment addressing: contiguous k order (k = 16*hi + s) in both A and B for
// V_WMMA_F32_16X16X32_BF16 (wave32).
//
// Register-pressure / codegen notes:
//  - B fragments loaded once per K-step; A fragments streamed one at a time
//    straight into their WMMAs (only one A frag live at a time).
//  - K-loop is 2-way unrolled so the scheduler double-buffers exactly one
//    K-step ahead (full unroll spilled; this fits in <256 VGPRs).
//  - Addresses are recomputed from the kernel-arg base each iteration so
//    infer-address-spaces keeps the loads GLOBAL (loop-carried pointer
//    arrays degraded them to FLAT, which drags DScnt into every wait).
//  - Next K-chunk prefetched via the speculative gfx1250 global_prefetch.
// ---------------------------------------------------------------------------
template <int MT, int NT>
__device__ __forceinline__ void wmma_tile(
    const bf16_t* __restrict__ A, const bf16_t* __restrict__ Bc,
    int K, int m0, int n0, int lane, v8f (&acc)[MT][NT]) {
  const int row = lane & 15;
  const int hi  = lane >> 4;   // 0 or 1 (lane half)
#pragma unroll 2
  for (int k0 = 0; k0 < K; k0 += 32) {
    v16bf bfrag[NT];
#pragma unroll
    for (int ni = 0; ni < NT; ++ni) {
      // B 32x16: lane column = N, 16 contiguous K values at 16*hi
      const bf16_t* bp = Bc + (size_t)(n0 + ni * 16 + row) * K + k0 + 16 * hi;
      v8bf lo = *(const v8bf*)bp;
      v8bf hh = *(const v8bf*)(bp + 8);
      __builtin_prefetch(bp + 32, 0, 1);   // speculative: safe past end
      bfrag[ni] = __builtin_shufflevector(lo, hh, 0, 1, 2, 3, 4, 5, 6, 7,
                                          8, 9, 10, 11, 12, 13, 14, 15);
    }
#pragma unroll
    for (int mi = 0; mi < MT; ++mi) {
      // A 16x32: lane row = M, 16 contiguous K values at 16*hi
      const bf16_t* ap = A + (size_t)(m0 + mi * 16 + row) * K + k0 + 16 * hi;
      v8bf lo = *(const v8bf*)ap;
      v8bf hh = *(const v8bf*)(ap + 8);
      __builtin_prefetch(ap + 32, 0, 1);
      v16bf afrag = __builtin_shufflevector(lo, hh, 0, 1, 2, 3, 4, 5, 6, 7,
                                            8, 9, 10, 11, 12, 13, 14, 15);
#pragma unroll
      for (int ni = 0; ni < NT; ++ni)
        acc[mi][ni] = __builtin_amdgcn_wmma_f32_16x16x32_bf16(
            false, afrag, false, bfrag[ni], (short)0, acc[mi][ni],
            false, false);
    }
  }
}

// Wave tiling shared by all GEMM kernels:
//   block = 256 threads = 8 wave32s, block tile 128(M) x 128(N),
//   wave tile 64(M) x 32(N): wave&1 -> M half, wave>>1 -> N quarter.
#define GEMM_WAVE_COORDS()                                   \
  int lane = threadIdx.x & 31, wave = threadIdx.x >> 5;      \
  int b = blockIdx.z;                                        \
  int m0 = blockIdx.x * 128 + (wave & 1) * 64;               \
  int n0 = blockIdx.y * 128 + (wave >> 1) * 32;              \
  int row = lane & 15, hi = lane >> 4;                       \
  (void)row; (void)hi;

// ---------------------------------------------------------------------------
// 1) Weight prep: bf16 conversion. W3 = [theta_w | phi_w | g_w] columns
//    (each is [Ci,C] row-major == column-major B with K=C), WB = W_w bf16,
//    bias3 = [theta_b | phi_b | g_b].
// ---------------------------------------------------------------------------
__global__ void k_prep(const float* __restrict__ g_w,
                       const float* __restrict__ theta_w,
                       const float* __restrict__ phi_w,
                       const float* __restrict__ W_w,
                       const float* __restrict__ g_b,
                       const float* __restrict__ theta_b,
                       const float* __restrict__ phi_b,
                       bf16_t* __restrict__ W3, bf16_t* __restrict__ WB,
                       float* __restrict__ bias3) {
  const int WSZ = CI * CC;  // 32768
  int i = blockIdx.x * blockDim.x + threadIdx.x;
  if (i < WSZ)               W3[i] = f2bf(theta_w[i]);
  else if (i < 2 * WSZ)      W3[i] = f2bf(phi_w[i - WSZ]);
  else if (i < 3 * WSZ)      W3[i] = f2bf(g_w[i - 2 * WSZ]);
  else if (i < 4 * WSZ)      WB[i - 3 * WSZ] = f2bf(W_w[i - 3 * WSZ]);
  if (i < CI)                bias3[i] = theta_b[i];
  else if (i < 2 * CI)       bias3[i] = phi_b[i - CI];
  else if (i < 3 * CI)       bias3[i] = g_b[i - 2 * CI];
}

// ---------------------------------------------------------------------------
// 2) x [B,C,N] f32 -> xT [B,N,C] bf16, LDS-tiled 32x32 transpose
// ---------------------------------------------------------------------------
__global__ void k_transpose_x(const float* __restrict__ x,
                              bf16_t* __restrict__ xT) {
  __shared__ float t[32][33];
  int b = blockIdx.z;
  int n0 = blockIdx.x * 32;
  int c0 = blockIdx.y * 32;
  int tx = threadIdx.x, ty = threadIdx.y;
#pragma unroll
  for (int j = 0; j < 4; ++j)
    t[ty + 8 * j][tx] =
        x[((size_t)(b * CC + c0 + ty + 8 * j)) * NN + n0 + tx];
  __syncthreads();
#pragma unroll
  for (int j = 0; j < 4; ++j)
    xT[((size_t)b * NN + n0 + ty + 8 * j) * CC + c0 + tx] =
        f2bf(t[tx][ty + 8 * j]);
}

// ---------------------------------------------------------------------------
// 3) Fused projection GEMM: [N,256] x [256 -> 384 cols] per batch.
//    cols 0..127 = theta (bf16, A-matrix of GEMM1), 128..255 = phi (f32),
//    256..383 = g (f32). Bias fused.
// ---------------------------------------------------------------------------
__global__ void k_gemm_proj(const bf16_t* __restrict__ xT,
                            const bf16_t* __restrict__ W3,
                            const float* __restrict__ bias3,
                            bf16_t* __restrict__ thetaA,
                            float* __restrict__ phiF,
                            float* __restrict__ gF) {
  GEMM_WAVE_COORDS();
  v8f acc[4][2] = {};
  wmma_tile<4, 2>(xT + (size_t)b * NN * CC, W3, CC, m0, n0, lane, acc);
#pragma unroll
  for (int mi = 0; mi < 4; ++mi)
#pragma unroll
    for (int ni = 0; ni < 2; ++ni) {
      int col = n0 + ni * 16 + row;
      float bias = bias3[col];
      int mb = m0 + mi * 16 + 8 * hi;
#pragma unroll
      for (int r = 0; r < 8; ++r) {
        float v = acc[mi][ni][r] + bias;
        int m = mb + r;
        if (col < CI)
          thetaA[((size_t)b * NN + m) * CI + col] = f2bf(v);
        else if (col < 2 * CI)
          phiF[((size_t)b * NN + m) * CI + (col - CI)] = v;
        else
          gF[((size_t)b * NN + m) * CI + (col - 2 * CI)] = v;
      }
    }
}

// ---------------------------------------------------------------------------
// 4) phi 2x2 maxpool: phiF[b][n][ci] f32 -> phiB[b][p][ci] bf16 (B of GEMM1)
// ---------------------------------------------------------------------------
__global__ void k_pool_phi(const float* __restrict__ phiF,
                           bf16_t* __restrict__ phiB) {
  int t = blockIdx.x * blockDim.x + threadIdx.x;  // B*NP*CI threads
  int ci = t & (CI - 1);
  int p = (t >> 7) & (NP - 1);
  int b = t >> 17;
  int ph = p >> 5, pw = p & 31;
  const float* base = phiF + (size_t)b * NN * CI;
  float v = -3.4e38f;
#pragma unroll
  for (int dh = 0; dh < 2; ++dh)
#pragma unroll
    for (int dw = 0; dw < 2; ++dw) {
      int n = (2 * ph + dh) * WW + 2 * pw + dw;
      v = fmaxf(v, base[(size_t)n * CI + ci]);
    }
  phiB[((size_t)b * NP + p) * CI + ci] = f2bf(v);
}

// ---------------------------------------------------------------------------
// 5) g 2x2 maxpool + transpose: gF[b][n][ci] -> gB[b][ci][p] bf16 (B of GEMM2)
// ---------------------------------------------------------------------------
__global__ void k_pool_g(const float* __restrict__ gF,
                         bf16_t* __restrict__ gB) {
  int t = blockIdx.x * blockDim.x + threadIdx.x;  // B*CI*NP threads
  int p = t & (NP - 1);
  int ci = (t >> 10) & (CI - 1);
  int b = t >> 17;
  int ph = p >> 5, pw = p & 31;
  const float* base = gF + (size_t)b * NN * CI;
  float v = -3.4e38f;
#pragma unroll
  for (int dh = 0; dh < 2; ++dh)
#pragma unroll
    for (int dw = 0; dw < 2; ++dw) {
      int n = (2 * ph + dh) * WW + 2 * pw + dw;
      v = fmaxf(v, base[(size_t)n * CI + ci]);
    }
  gB[((size_t)b * CI + ci) * NP + p] = f2bf(v);
}

// ---------------------------------------------------------------------------
// 6) Attention score GEMM: f[b] = thetaA[4096,128] x phiB -> [4096,1024] bf16
// ---------------------------------------------------------------------------
__global__ void k_gemm_f(const bf16_t* __restrict__ thetaA,
                         const bf16_t* __restrict__ phiB,
                         bf16_t* __restrict__ fP) {
  GEMM_WAVE_COORDS();
  v8f acc[4][2] = {};
  wmma_tile<4, 2>(thetaA + (size_t)b * NN * CI,
                  phiB + (size_t)b * NP * CI, CI, m0, n0, lane, acc);
#pragma unroll
  for (int mi = 0; mi < 4; ++mi)
#pragma unroll
    for (int ni = 0; ni < 2; ++ni) {
      int col = n0 + ni * 16 + row;
      int mb = m0 + mi * 16 + 8 * hi;
#pragma unroll
      for (int r = 0; r < 8; ++r)
        fP[((size_t)b * NN + mb + r) * NP + col] = f2bf(acc[mi][ni][r]);
    }
}

// ---------------------------------------------------------------------------
// 7) Row softmax over 1024 cols, in-place on bf16 scores (f32 math)
// ---------------------------------------------------------------------------
__global__ void k_softmax(bf16_t* __restrict__ fP) {
  __shared__ float red[256];
  size_t rowi = blockIdx.x;  // B*NN rows
  bf16_t* rp = fP + rowi * NP;
  int t = threadIdx.x;
  float v[4];
  float mx = -3.4e38f;
#pragma unroll
  for (int i = 0; i < 4; ++i) {
    v[i] = bf2f(rp[t + 256 * i]);
    mx = fmaxf(mx, v[i]);
  }
  red[t] = mx;
  __syncthreads();
  for (int s = 128; s > 0; s >>= 1) {
    if (t < s) red[t] = fmaxf(red[t], red[t + s]);
    __syncthreads();
  }
  mx = red[0];
  __syncthreads();
  float sum = 0.f;
#pragma unroll
  for (int i = 0; i < 4; ++i) {
    v[i] = __expf(v[i] - mx);
    sum += v[i];
  }
  red[t] = sum;
  __syncthreads();
  for (int s = 128; s > 0; s >>= 1) {
    if (t < s) red[t] += red[t + s];
    __syncthreads();
  }
  float inv = 1.f / red[0];
#pragma unroll
  for (int i = 0; i < 4; ++i) rp[t + 256 * i] = f2bf(v[i] * inv);
}

// ---------------------------------------------------------------------------
// 8) y GEMM: y[b] = P[4096,1024] x gB -> [4096,128] bf16
// ---------------------------------------------------------------------------
__global__ void k_gemm_y(const bf16_t* __restrict__ fP,
                         const bf16_t* __restrict__ gB,
                         bf16_t* __restrict__ yB) {
  GEMM_WAVE_COORDS();
  v8f acc[4][2] = {};
  wmma_tile<4, 2>(fP + (size_t)b * NN * NP,
                  gB + (size_t)b * CI * NP, NP, m0, n0, lane, acc);
#pragma unroll
  for (int mi = 0; mi < 4; ++mi)
#pragma unroll
    for (int ni = 0; ni < 2; ++ni) {
      int col = n0 + ni * 16 + row;
      int mb = m0 + mi * 16 + 8 * hi;
#pragma unroll
      for (int r = 0; r < 8; ++r)
        yB[((size_t)b * NN + mb + r) * CI + col] = f2bf(acc[mi][ni][r]);
    }
}

// ---------------------------------------------------------------------------
// 9) W conv GEMM + fused bias + BatchNorm + residual: out f32 [B,256,4096].
//    Each lane's 8 D values are 8 consecutive floats along N -> store as
//    two b128s; residual x loaded the same way.
// ---------------------------------------------------------------------------
__global__ void k_gemm_out(const bf16_t* __restrict__ yB,
                           const bf16_t* __restrict__ WB,
                           const float* __restrict__ W_b,
                           const float* __restrict__ bn_gamma,
                           const float* __restrict__ bn_beta,
                           const float* __restrict__ bn_mean,
                           const float* __restrict__ bn_var,
                           const float* __restrict__ x,
                           float* __restrict__ out) {
  GEMM_WAVE_COORDS();
  v8f acc[4][2] = {};
  wmma_tile<4, 2>(yB + (size_t)b * NN * CI, WB, CI, m0, n0, lane, acc);
#pragma unroll
  for (int mi = 0; mi < 4; ++mi)
#pragma unroll
    for (int ni = 0; ni < 2; ++ni) {
      int col = n0 + ni * 16 + row;  // output channel o
      float inv = bn_gamma[col] * rsqrtf(bn_var[col] + EPSV);
      float add = (W_b[col] - bn_mean[col]) * inv + bn_beta[col];
      int mb = m0 + mi * 16 + 8 * hi;
      size_t idx = ((size_t)(b * CC + col)) * NN + mb;
      float4 xa = *(const float4*)(x + idx);
      float4 xb = *(const float4*)(x + idx + 4);
      float4 o0, o1;
      o0.x = acc[mi][ni][0] * inv + add + xa.x;
      o0.y = acc[mi][ni][1] * inv + add + xa.y;
      o0.z = acc[mi][ni][2] * inv + add + xa.z;
      o0.w = acc[mi][ni][3] * inv + add + xa.w;
      o1.x = acc[mi][ni][4] * inv + add + xb.x;
      o1.y = acc[mi][ni][5] * inv + add + xb.y;
      o1.z = acc[mi][ni][6] * inv + add + xb.z;
      o1.w = acc[mi][ni][7] * inv + add + xb.w;
      *(float4*)(out + idx) = o0;
      *(float4*)(out + idx + 4) = o1;
    }
}

// ---------------------------------------------------------------------------
extern "C" void kernel_launch(void* const* d_in, const int* in_sizes, int n_in,
                              void* d_out, int out_size, void* d_ws,
                              size_t ws_size, hipStream_t stream) {
  const float* x        = (const float*)d_in[0];
  const float* g_w      = (const float*)d_in[1];
  const float* g_b      = (const float*)d_in[2];
  const float* theta_w  = (const float*)d_in[3];
  const float* theta_b  = (const float*)d_in[4];
  const float* phi_w    = (const float*)d_in[5];
  const float* phi_b    = (const float*)d_in[6];
  const float* W_w      = (const float*)d_in[7];
  const float* W_b      = (const float*)d_in[8];
  const float* bn_gamma = (const float*)d_in[9];
  const float* bn_beta  = (const float*)d_in[10];
  const float* bn_mean  = (const float*)d_in[11];
  const float* bn_var   = (const float*)d_in[12];
  float* out = (float*)d_out;

  char* p = (char*)d_ws;
  auto alloc = [&](size_t bytes) {
    char* r = p;
    p += (bytes + 255) & ~(size_t)255;
    return r;
  };
  bf16_t* xT     = (bf16_t*)alloc((size_t)BN * NN * CC * 2);  // 16.8 MB
  bf16_t* thetaA = (bf16_t*)alloc((size_t)BN * NN * CI * 2);  //  8.4 MB
  float*  phiF   = (float*) alloc((size_t)BN * NN * CI * 4);  // 16.8 MB
  float*  gF     = (float*) alloc((size_t)BN * NN * CI * 4);  // 16.8 MB
  bf16_t* phiB   = (bf16_t*)alloc((size_t)BN * NP * CI * 2);  //  2.1 MB
  bf16_t* gB     = (bf16_t*)alloc((size_t)BN * CI * NP * 2);  //  2.1 MB
  bf16_t* fP     = (bf16_t*)alloc((size_t)BN * NN * NP * 2);  // 67.1 MB
  bf16_t* yB     = (bf16_t*)alloc((size_t)BN * NN * CI * 2);  //  8.4 MB
  bf16_t* W3     = (bf16_t*)alloc((size_t)3 * CI * CC * 2);
  bf16_t* WB     = (bf16_t*)alloc((size_t)CC * CI * 2);
  float*  bias3  = (float*) alloc((size_t)3 * CI * 4);

  k_prep<<<512, 256, 0, stream>>>(g_w, theta_w, phi_w, W_w, g_b, theta_b,
                                  phi_b, W3, WB, bias3);
  k_transpose_x<<<dim3(NN / 32, CC / 32, BN), dim3(32, 8), 0, stream>>>(x, xT);
  k_gemm_proj<<<dim3(NN / 128, 384 / 128, BN), 256, 0, stream>>>(
      xT, W3, bias3, thetaA, phiF, gF);
  k_pool_phi<<<(BN * NP * CI) / 256, 256, 0, stream>>>(phiF, phiB);
  k_pool_g<<<(BN * CI * NP) / 256, 256, 0, stream>>>(gF, gB);
  k_gemm_f<<<dim3(NN / 128, NP / 128, BN), 256, 0, stream>>>(thetaA, phiB, fP);
  k_softmax<<<BN * NN, 256, 0, stream>>>(fP);
  k_gemm_y<<<dim3(NN / 128, CI / 128, BN), 256, 0, stream>>>(fP, gB, yB);
  k_gemm_out<<<dim3(NN / 128, CC / 128, BN), 256, 0, stream>>>(
      yB, WB, W_b, bn_gamma, bn_beta, bn_mean, bn_var, x, out);
}
